// BiLSTM_CRF_20512763806304
// MI455X (gfx1250) — matrix operand order repaired and checked by reading, not compiled, not run on hardware
//
#include <hip/hip_runtime.h>
#include <hip/hip_bf16.h>

// ---- problem constants (from reference setup_inputs) ----
#define B_   32
#define S_   512
#define E_   300
#define KPAD 320          // E padded to multiple of 32 (WMMA K-tile)
#define HD   256
#define G_   1024         // 4*HD
#define H2   512          // 2*HD
#define T_   11
#define START_TAG 9
#define STOP_TAG  10
#define HPAD 264          // LDS row stride for h (f16), padded vs 256

typedef _Float16 f16;
typedef __attribute__((ext_vector_type(16))) _Float16 v16h;
typedef __attribute__((ext_vector_type(8)))  _Float16 v8h;
typedef __attribute__((ext_vector_type(8)))  float    v8f;

__device__ __forceinline__ float sigm(float x) { return 1.0f / (1.0f + __expf(-x)); }

// ===================== prep: embedding gather -> f16, K padded =====================
__global__ void embed_gather(const int* __restrict__ data, const float* __restrict__ emb,
                             f16* __restrict__ x) {
  size_t idx = (size_t)blockIdx.x * blockDim.x + threadIdx.x;
  size_t total = (size_t)S_ * B_ * KPAD;
  if (idx >= total) return;
  int kk = (int)(idx % KPAD);
  int m  = (int)(idx / KPAD);         // m = s*B + b  (x layout is (S,B,E))
  int s = m / B_, b = m % B_;
  int tok = data[b * S_ + s];         // data is (B,S)
  float v = (kk < E_) ? emb[(size_t)tok * E_ + kk] : 0.0f;
  x[idx] = (f16)v;
}

// ===================== prep: f32 weight -> f16, optional K padding =====================
__global__ void cvt_pad(const float* __restrict__ w, f16* __restrict__ o,
                        int rows, int kin, int kout) {
  size_t idx = (size_t)blockIdx.x * blockDim.x + threadIdx.x;
  size_t total = (size_t)rows * kout;
  if (idx >= total) return;
  int kk = (int)(idx % kout);
  int r  = (int)(idx / kout);
  o[idx] = (kk < kin) ? (f16)w[(size_t)r * kin + kk] : (f16)0.0f;
}

// ===================== input projection: xw = x @ w_ih^T + b  (both dirs) =============
// grid = (M/16 = 1024, 4), block = 256 (8 waves). Each wave: 1 M-tile x 4 N-tiles.
__global__ __launch_bounds__(256) void gemm_xw(const f16* __restrict__ x,
                                               const f16* __restrict__ wih_all,
                                               const float* __restrict__ bf,
                                               const float* __restrict__ bb,
                                               float* __restrict__ xw_all) {
  int tid = threadIdx.x;
  int wave = tid >> 5, lane = tid & 31;
  int laneN = lane & 15, laneHi = lane >> 4;
  int ntile0 = (blockIdx.y * 8 + wave) * 4;   // 0..124 over 2*64 N-tiles
  int dir   = ntile0 >> 6;
  int nloc0 = ntile0 & 63;
  const f16* wih = wih_all + (size_t)dir * G_ * KPAD;   // (G, KPAD) row-major == B^T
  const float* bias = dir ? bb : bf;
  int mrow = blockIdx.x * 16 + laneN;

  v8f acc[4] = {};
  const f16* xrow = x + (size_t)mrow * KPAD;
  #pragma unroll
  for (int k0 = 0; k0 < KPAD; k0 += 32) {
    // A fragment (16x32 f16): lane holds 16 consecutive K at (lane/16)*16
    v8h lo = *(const v8h*)(xrow + k0 + laneHi * 16);
    v8h hi = *(const v8h*)(xrow + k0 + 8 + laneHi * 16);
    v16h a;
    #pragma unroll
    for (int i = 0; i < 8; ++i) { a[i] = lo[i]; a[i + 8] = hi[i]; }
    #pragma unroll
    for (int q = 0; q < 4; ++q) {
      // B fragment (32x16): lane holds column n=lane%16, 16 consecutive K at (lane/16)*16
      int grow = (nloc0 + q) * 16 + laneN;
      v16h bfr = *(const v16h*)(wih + (size_t)grow * KPAD + k0 + laneHi * 16);
      acc[q] = __builtin_amdgcn_wmma_f32_16x16x32_f16(false, a, false, bfr,
                                                      (short)0, acc[q], false, false);
    }
  }
  float* xw = xw_all + (size_t)dir * S_ * B_ * G_;
  #pragma unroll
  for (int q = 0; q < 4; ++q) {
    #pragma unroll
    for (int r = 0; r < 8; ++r) {
      int M = blockIdx.x * 16 + r + 8 * laneHi;
      int n = (nloc0 + q) * 16 + laneN;
      xw[(size_t)M * G_ + n] = acc[q][r] + bias[n];
    }
  }
}

// ===================== persistent LSTM recurrence: one WGP per direction ==============
// block = 1024 (32 waves). wave = (mtile:1)(j:4). Wave owns hidden cols j*16..+15 for
// ALL FOUR gates (N = g*256 + j*16), so gate math runs on in-register WMMA fragments.
__global__ __launch_bounds__(1024) void lstm_rec(const f16* __restrict__ whh_all,
                                                 const float* __restrict__ xw_all,
                                                 float* __restrict__ h_all) {
  __shared__ __align__(32) f16 h_sh[32][HPAD];
  int dir = blockIdx.x;
  const f16* whh = whh_all + (size_t)dir * G_ * HD;
  const float* xw = xw_all + (size_t)dir * S_ * B_ * G_;
  int tid = threadIdx.x;
  int wave = tid >> 5, lane = tid & 31;
  int laneN = lane & 15, laneHi = lane >> 4;
  int mtile = wave >> 4;          // 0..1  (batch tile)
  int j = wave & 15;              // 0..15 (hidden-column tile within each gate block)

  for (int idx = tid; idx < 32 * HPAD; idx += 1024) ((f16*)h_sh)[idx] = (f16)0.0f;
  v8f cfrag = {};                 // cell state fragment, lives in registers
  __syncthreads();

  for (int step = 0; step < S_; ++step) {
    int t = dir ? (S_ - 1 - step) : step;
    v8f acc[4] = {};
    #pragma unroll
    for (int k0 = 0; k0 < HD; k0 += 32) {
      const f16* hp = &h_sh[mtile * 16 + laneN][k0 + laneHi * 16];
      v8h lo = *(const v8h*)(hp);
      v8h hi = *(const v8h*)(hp + 8);
      v16h a;
      #pragma unroll
      for (int i = 0; i < 8; ++i) { a[i] = lo[i]; a[i + 8] = hi[i]; }
      #pragma unroll
      for (int g = 0; g < 4; ++g) {
        int grow = g * HD + j * 16 + laneN;     // w_hh row (N,K layout) - L2 resident
        v16h bfr = *(const v16h*)(whh + (size_t)grow * HD + k0 + laneHi * 16);
        acc[g] = __builtin_amdgcn_wmma_f32_16x16x32_f16(false, a, false, bfr,
                                                        (short)0, acc[g], false, false);
      }
    }
    __syncthreads();   // all waves done reading previous h before overwrite
    const float* xwb = xw + (size_t)t * B_ * G_;
    #pragma unroll
    for (int r = 0; r < 8; ++r) {
      int M = mtile * 16 + r + 8 * laneHi;      // batch index
      int n = j * 16 + laneN;                   // hidden column
      const float* xr = xwb + (size_t)M * G_;
      float gi = acc[0][r] + xr[0 * HD + n];
      float gf = acc[1][r] + xr[1 * HD + n];
      float gg = acc[2][r] + xr[2 * HD + n];
      float go = acc[3][r] + xr[3 * HD + n];
      float c = sigm(gf) * cfrag[r] + sigm(gi) * tanhf(gg);
      float h = sigm(go) * tanhf(c);
      cfrag[r] = c;
      h_sh[M][n] = (f16)h;
      h_all[((size_t)t * B_ + M) * H2 + dir * HD + n] = h;
    }
    __syncthreads();   // new h visible to all waves for next step
  }
}

// ===================== feats[b,s,t] = h[s,b,:] . w_tag[t,:] + b_tag[t] ================
__global__ void feats_kernel(const float* __restrict__ h_all, const float* __restrict__ w_tag,
                             const float* __restrict__ b_tag, float* __restrict__ feats) {
  int idx = blockIdx.x * blockDim.x + threadIdx.x;
  if (idx >= B_ * S_ * T_) return;
  int t = idx % T_;
  int bs = idx / T_;
  int s = bs % S_;
  int b = bs / S_;
  const float* h = h_all + ((size_t)s * B_ + b) * H2;
  const float* w = w_tag + (size_t)t * H2;
  float acc = 0.0f;
  for (int k = 0; k < H2; k += 4)
    acc += h[k] * w[k] + h[k + 1] * w[k + 1] + h[k + 2] * w[k + 2] + h[k + 3] * w[k + 3];
  feats[((size_t)b * S_ + s) * T_ + t] = acc + b_tag[t];
}

// ===================== CRF forward + gold score + loss (one wave per batch) ===========
__global__ __launch_bounds__(1024) void crf_kernel(const float* __restrict__ feats,
                                                   const int* __restrict__ label,
                                                   const int* __restrict__ lengths,
                                                   const float* __restrict__ trans,
                                                   float* __restrict__ out) {
  __shared__ float trans_s[T_ * T_];
  __shared__ float loss_sh[32];
  int tid = threadIdx.x, wave = tid >> 5, lane = tid & 31;
  if (tid < T_ * T_) trans_s[tid] = trans[tid];
  __syncthreads();

  int b = wave;
  const float* fb = feats + (size_t)b * S_ * T_;
  const int* lab = label + b * S_;
  int len = lengths[b];
  int col = (lane < T_) ? lane : 0;             // keep all lanes active for shfl
  float prev = (lane < T_) ? (fb[lane] + trans_s[START_TAG * T_ + lane]) : -1e30f;

  for (int t = 1; t < S_; ++t) {
    float vals[T_];
    float m = -1e30f;
    #pragma unroll
    for (int i = 0; i < T_; ++i) {
      float pi = __shfl(prev, i, 32);
      float v = pi + trans_s[i * T_ + col];
      vals[i] = v;
      m = fmaxf(m, v);
    }
    float ss = 0.0f;
    #pragma unroll
    for (int i = 0; i < T_; ++i) ss += __expf(vals[i] - m);
    float cand = m + __logf(ss) + fb[(size_t)t * T_ + col];
    if (lane < T_ && t < len) prev = cand;
  }
  // forward score = logsumexp over tags
  float m = -1e30f;
  #pragma unroll
  for (int i = 0; i < T_; ++i) m = fmaxf(m, __shfl(prev, i, 32));
  float ss = 0.0f;
  #pragma unroll
  for (int i = 0; i < T_; ++i) ss += __expf(__shfl(prev, i, 32) - m);
  float fscore = m + __logf(ss);

  // gold score: lanes stride over sequence, wave-reduce
  float emit = 0.0f, trs = 0.0f;
  for (int s = lane; s < S_; s += 32) {
    if (s < len) {
      emit += fb[(size_t)s * T_ + lab[s]];
      if (s >= 1) trs += trans_s[lab[s - 1] * T_ + lab[s]];
    }
  }
  #pragma unroll
  for (int off = 16; off > 0; off >>= 1) {
    emit += __shfl_down(emit, off, 32);
    trs  += __shfl_down(trs,  off, 32);
  }
  if (lane == 0) {
    float gold = emit + trs + trans_s[START_TAG * T_ + lab[0]]
               + trans_s[lab[len - 1] * T_ + STOP_TAG];
    loss_sh[wave] = fscore - gold;
  }
  __syncthreads();
  if (tid == 0) {
    float tot = 0.0f;
    for (int w = 0; w < 32; ++w) tot += loss_sh[w];
    out[0] = tot / (float)B_;
  }
}

// ===================== launch =====================
extern "C" void kernel_launch(void* const* d_in, const int* in_sizes, int n_in,
                              void* d_out, int out_size, void* d_ws, size_t ws_size,
                              hipStream_t stream) {
  const int*   data  = (const int*)  d_in[0];
  const int*   label = (const int*)  d_in[1];
  const int*   lens  = (const int*)  d_in[2];
  const float* emb   = (const float*)d_in[3];
  const float* wih_f = (const float*)d_in[4];
  const float* whh_f = (const float*)d_in[5];
  const float* b_f   = (const float*)d_in[6];
  const float* wih_b = (const float*)d_in[7];
  const float* whh_b = (const float*)d_in[8];
  const float* b_b   = (const float*)d_in[9];
  const float* w_tag = (const float*)d_in[10];
  const float* b_tag = (const float*)d_in[11];
  const float* trans = (const float*)d_in[12];

  char* ws = (char*)d_ws;
  size_t off = 0;
  auto alloc = [&](size_t bytes) { size_t o = off; off += (bytes + 255) & ~(size_t)255; return o; };
  size_t o_x   = alloc((size_t)S_ * B_ * KPAD * sizeof(f16));   // x as f16 (S,B,KPAD)
  size_t o_wih = alloc((size_t)2 * G_ * KPAD * sizeof(f16));    // w_ih f16, both dirs
  size_t o_whh = alloc((size_t)2 * G_ * HD   * sizeof(f16));    // w_hh f16, both dirs
  size_t o_xw  = alloc((size_t)2 * S_ * B_ * G_ * sizeof(float)); // xw, both dirs
  size_t o_h   = alloc((size_t)S_ * B_ * H2 * sizeof(float));   // h concat (S,B,2Hd)
  size_t o_f   = alloc((size_t)B_ * S_ * T_ * sizeof(float));   // feats (B,S,T)

  f16*   x16   = (f16*)(ws + o_x);
  f16*   wih16 = (f16*)(ws + o_wih);
  f16*   whh16 = (f16*)(ws + o_whh);
  float* xw    = (float*)(ws + o_xw);
  float* hall  = (float*)(ws + o_h);
  float* feat  = (float*)(ws + o_f);

  { size_t n = (size_t)S_ * B_ * KPAD;
    embed_gather<<<(unsigned)((n + 255) / 256), 256, 0, stream>>>(data, emb, x16); }
  { size_t n = (size_t)G_ * KPAD;
    cvt_pad<<<(unsigned)((n + 255) / 256), 256, 0, stream>>>(wih_f, wih16,                 G_, E_, KPAD);
    cvt_pad<<<(unsigned)((n + 255) / 256), 256, 0, stream>>>(wih_b, wih16 + (size_t)G_*KPAD, G_, E_, KPAD); }
  { size_t n = (size_t)G_ * HD;
    cvt_pad<<<(unsigned)((n + 255) / 256), 256, 0, stream>>>(whh_f, whh16,                G_, HD, HD);
    cvt_pad<<<(unsigned)((n + 255) / 256), 256, 0, stream>>>(whh_b, whh16 + (size_t)G_*HD, G_, HD, HD); }

  gemm_xw<<<dim3(S_ * B_ / 16, 4), 256, 0, stream>>>(x16, wih16, b_f, b_b, xw);
  lstm_rec<<<2, 1024, 0, stream>>>(whh16, xw, hall);

  { size_t n = (size_t)B_ * S_ * T_;
    feats_kernel<<<(unsigned)((n + 255) / 256), 256, 0, stream>>>(hall, w_tag, b_tag, feat); }
  crf_kernel<<<1, 1024, 0, stream>>>(feat, label, lens, trans, (float*)d_out);

  (void)in_sizes; (void)n_in; (void)out_size; (void)ws_size;
}
